// MoELayer_68186900791359
// MI455X (gfx1250) — matrix operand order repaired and checked by reading, not compiled, not run on hardware
//
#include <hip/hip_runtime.h>
#include <hip/hip_bf16.h>

// ---------------------------------------------------------------------------
// MoE (T=8192, d=1024, dff=1024, E=64, k=2, capacity=512) for gfx1250.
// bf16 WMMA (v_wmma_f32_16x16x32_bf16) grouped GEMMs, fused SwiGLU.
// A-tiles staged via Tensor Data Mover (tensor_load_to_lds) when available.
// Workspace layout (needs ~257 MB):
//   pos[N] int | xd[E*C*d] bf16 | act[E*C*dff] bf16 | yd[E*C*d] f32
// ---------------------------------------------------------------------------

typedef __attribute__((ext_vector_type(16))) __bf16 v16bf;
typedef __attribute__((ext_vector_type(8)))  float  v8f;

union Frag16 { uint4 q[2]; v16bf v; };

#if defined(__has_builtin)
#  if __has_builtin(__builtin_amdgcn_cvt_pk_bf16_f32)
#    define HAVE_CVT_PK_BF16 1
#  endif
#  if __has_builtin(__builtin_amdgcn_tensor_load_to_lds) && \
      __has_builtin(__builtin_amdgcn_s_wait_tensorcnt)
#    define HAVE_TDM 1
#  endif
#endif

// Pack two f32 into a u32 holding {bf16(a) low, bf16(b) high}.
// Fallback: round-to-nearest (ties away) = one add per element + one v_perm.
__device__ __forceinline__ unsigned pack2bf(float a, float b) {
#ifdef HAVE_CVT_PK_BF16
  auto r = __builtin_amdgcn_cvt_pk_bf16_f32(a, b);   // v_cvt_pk_bf16_f32
  unsigned u;
  __builtin_memcpy(&u, &r, 4);
  return u;
#else
  union { float f; unsigned u; } xa, xb; xa.f = a; xb.f = b;
  return __builtin_amdgcn_perm(xb.u + 0x8000u, xa.u + 0x8000u, 0x07060302u);
#endif
}

__device__ __forceinline__ unsigned short f2bf1(float a) {
  union { float f; unsigned u; } x; x.f = a;
  return (unsigned short)((x.u + 0x8000u) >> 16);
}

#ifdef HAVE_TDM
typedef __attribute__((ext_vector_type(4))) unsigned int uv4;
typedef __attribute__((ext_vector_type(8))) int          iv8;
typedef __attribute__((ext_vector_type(4))) int          iv4;

// TDM copy of a 128x32 bf16 tile (row stride `stride` elements) into LDS.
// D# per CDNA5 ISA ch.8: group0 = {count, lds_addr, global_addr, type=2},
// group1 = {data_size=2B, tensor dims (large), tile 32x128, dim0 stride}.
__device__ __forceinline__ void tdm_load_tile_128x32_bf16(
    const unsigned short* gsrc, unsigned short* lds_dst, int stride) {
  unsigned lds = (unsigned)(uintptr_t)lds_dst;
  unsigned long long ga = (unsigned long long)(uintptr_t)gsrc;
  uv4 g0;
  g0[0] = 1u;                                   // count=1, no gather
  g0[1] = lds;                                  // lds_addr (bytes)
  g0[2] = (unsigned)ga;                         // global_addr[31:0]
  g0[3] = (unsigned)(ga >> 32) | (2u << 30);    // global_addr[56:32] | type=2
  iv8 g1;
  g1[0] = 1 << 16;                              // data_size=1 (2 bytes)
  g1[1] = 0;                                    // tensor_dim0=1M -> low16=0
  g1[2] = 0x10;                                 // dim0 hi | tensor_dim1=1M lo
  g1[3] = 0x10 | (32 << 16);                    // dim1 hi | tile_dim0=32
  g1[4] = 128;                                  // tile_dim1=128, tile_dim2=0
  g1[5] = stride;                               // tensor_dim0_stride lo32
  g1[6] = 0;
  g1[7] = 0;
  iv4 gz = {0, 0, 0, 0};
#if __clang_major__ >= 23
  iv8 gz8 = {0, 0, 0, 0, 0, 0, 0, 0};
  __builtin_amdgcn_tensor_load_to_lds(g0, g1, gz, gz, gz8, 0);
#else
  __builtin_amdgcn_tensor_load_to_lds(g0, g1, gz, gz, 0);
#endif
}
#endif  // HAVE_TDM

constexpr int T_   = 8192;
constexpr int D_   = 1024;
constexpr int DFF_ = 1024;
constexpr int E_   = 64;
constexpr int K_   = 2;
constexpr int N_   = T_ * K_;     // 16384 expanded tokens
constexpr int CAP_ = 512;         // (2*N)/E

// ------------------------------ routing ------------------------------------
__global__ void route_kernel(const int* __restrict__ idx, int* __restrict__ pos) {
  int e = threadIdx.x;            // 64 threads
  int count = 0;
  for (int n = 0; n < N_; ++n) {
    if (idx[n] == e) pos[n] = count++;
  }
}

// ------------------------------ zero xd ------------------------------------
__global__ void zero_xd_kernel(uint4* __restrict__ p) {
  size_t i = (size_t)blockIdx.x * blockDim.x + threadIdx.x;
  p[i] = make_uint4(0u, 0u, 0u, 0u);
}

// ------------------------------ dispatch -----------------------------------
__global__ void dispatch_kernel(const float* __restrict__ hs,
                                const int* __restrict__ idx,
                                const int* __restrict__ pos,
                                unsigned short* __restrict__ xd) {
  int n = blockIdx.x;
  int p = pos[n];
  if (p >= CAP_) return;          // capacity drop
  int e   = idx[n];
  int tok = n >> 1;               // K_ == 2
  float4 f = ((const float4*)(hs + (size_t)tok * D_))[threadIdx.x];
  uint2 o;
  o.x = pack2bf(f.x, f.y);
  o.y = pack2bf(f.z, f.w);
  ((uint2*)(xd + ((size_t)e * CAP_ + p) * D_))[threadIdx.x] = o;
}

// --------------------- GEMM1 + SwiGLU (xd @ gate_up) -----------------------
// Block tile: 128 rows x (64 gate cols + paired 64 up cols), BK=32.
// 8 wave32s in a 4(m) x 2(n) grid; each wave owns 32x32 gate + 32x32 up.
__global__ void __launch_bounds__(256)
gemm1_swiglu_kernel(const unsigned short* __restrict__ xd,
                    const float* __restrict__ gate_up,
                    unsigned short* __restrict__ act) {
  constexpr int BK = 32;
  const int e    = blockIdx.z;
  const int mt   = blockIdx.x;    // 0..3   (rows of C=512)
  const int nt   = blockIdx.y;    // 0..15  (64-col slab of dff=1024)
  const int tid  = threadIdx.x;
  const int lane = tid & 31;
  const int wid  = tid >> 5;
  const int wm   = wid & 3;
  const int wn   = wid >> 2;

  __shared__ unsigned short sA[128 * BK];   // row-major, stride BK
  __shared__ unsigned short sBg[BK * 64];   // WMMA-B fragment layout
  __shared__ unsigned short sBu[BK * 64];

  const unsigned short* Abase = xd + ((size_t)e * CAP_ + (size_t)mt * 128) * D_;
  const float* Bg = gate_up + (size_t)e * D_ * (2 * DFF_) + nt * 64;
  const float* Bu = Bg + DFF_;

  const v8f vzero = {0.f, 0.f, 0.f, 0.f, 0.f, 0.f, 0.f, 0.f};
  v8f accG[2][2], accU[2][2];
#pragma unroll
  for (int i = 0; i < 2; ++i)
#pragma unroll
    for (int j = 0; j < 2; ++j) { accG[i][j] = vzero; accU[i][j] = vzero; }

  // Manual A loader mapping (fallback path)
  const int aRow = tid >> 1, aHalf = tid & 1;
  // B loader: 128 threads per matrix; each thread handles TWO adjacent K rows
  // x 8 cols, so each bf16 pair lands in adjacent fragment slots (b32 store).
  const int bSel  = tid >> 7;        // 0 = gate, 1 = up
  const int btt   = tid & 127;
  const int bKr   = (btt >> 3) * 2;  // even K row: 0,2,..,30
  const int bN0   = (btt & 7) * 8;   // 0..56
  const int bNb   = bN0 >> 4;        // 16-col fragment index
  const int bNlo  = bN0 & 15;        // 0 or 8
  const int bLsel = bKr & 16;        // lanes 16-31 hold K>=16
  const int bSlot = bKr & 15;        // even slot; pair fills slot, slot+1

  for (int k0 = 0; k0 < D_; k0 += BK) {
    __syncthreads();
#ifdef HAVE_TDM
    if (wid == 0) tdm_load_tile_128x32_bf16(Abase + k0, sA, D_);
#else
    { // stage A (bf16, row-major)
      const uint4* s = (const uint4*)(Abase + (size_t)aRow * D_ + k0 + aHalf * 16);
      uint4 a0 = s[0], a1 = s[1];
      uint4* d = (uint4*)&sA[aRow * BK + aHalf * 16];
      d[0] = a0; d[1] = a1;
    }
#endif
    { // stage B gate+up: f32 -> packed bf16 pairs into WMMA B-fragment layout
      const float* s0 = (bSel ? Bu : Bg) + (size_t)(k0 + bKr) * (2 * DFF_) + bN0;
      const float* s1 = s0 + 2 * DFF_;                 // next K row
      unsigned short* dst = bSel ? sBu : sBg;
      float4 r0a = ((const float4*)s0)[0];
      float4 r0b = ((const float4*)s0)[1];
      float4 r1a = ((const float4*)s1)[0];
      float4 r1b = ((const float4*)s1)[1];
      float v0[8] = {r0a.x, r0a.y, r0a.z, r0a.w, r0b.x, r0b.y, r0b.z, r0b.w};
      float v1[8] = {r1a.x, r1a.y, r1a.z, r1a.w, r1b.x, r1b.y, r1b.z, r1b.w};
      unsigned short* base = dst + bNb * 512 + bSlot;
#pragma unroll
      for (int j = 0; j < 8; ++j) {
        unsigned pk = pack2bf(v0[j], v1[j]);           // {K=k, K=k+1}
        *(unsigned*)&base[(bNlo + j + bLsel) * 16] = pk;
      }
    }
#ifdef HAVE_TDM
    if (wid == 0) __builtin_amdgcn_s_wait_tensorcnt(0);
#endif
    __syncthreads();

    if (k0 + BK < D_) {   // global_prefetch_b8 of next tiles
      __builtin_prefetch(Abase + (size_t)aRow * D_ + k0 + BK, 0, 1);
      __builtin_prefetch((bSel ? Bu : Bg) + (size_t)(k0 + BK + bKr) * (2 * DFF_) + bN0, 0, 1);
    }

    // A fragments (contiguous k = 16*(lane/16) + s, same order as B)
    Frag16 a[2];
#pragma unroll
    for (int mf = 0; mf < 2; ++mf) {
      int r  = wm * 32 + mf * 16 + (lane & 15);
      int kh = lane >> 4;
      a[mf].q[0] = *(const uint4*)&sA[r * BK + kh * 16];
      a[mf].q[1] = *(const uint4*)&sA[r * BK + kh * 16 + 8];
    }
    // B fragments: contiguous 32B per lane thanks to the staged layout
    Frag16 bg[2], bu[2];
#pragma unroll
    for (int nf = 0; nf < 2; ++nf) {
      int nb = wn * 2 + nf;
      const uint4* pg = (const uint4*)&sBg[nb * 512 + lane * 16];
      bg[nf].q[0] = pg[0]; bg[nf].q[1] = pg[1];
      const uint4* pu = (const uint4*)&sBu[nb * 512 + lane * 16];
      bu[nf].q[0] = pu[0]; bu[nf].q[1] = pu[1];
    }
#pragma unroll
    for (int mf = 0; mf < 2; ++mf)
#pragma unroll
      for (int nf = 0; nf < 2; ++nf) {
        accG[mf][nf] = __builtin_amdgcn_wmma_f32_16x16x32_bf16(
            false, a[mf].v, false, bg[nf].v, (short)0, accG[mf][nf], false, false);
        accU[mf][nf] = __builtin_amdgcn_wmma_f32_16x16x32_bf16(
            false, a[mf].v, false, bu[nf].v, (short)0, accU[mf][nf], false, false);
      }
  }

  // Fused SwiGLU epilogue -> act (bf16)
  const int mrow0 = mt * 128 + wm * 32;
  const int ncol0 = nt * 64 + wn * 32;
  unsigned short* actE = act + (size_t)e * CAP_ * DFF_;
#pragma unroll
  for (int mf = 0; mf < 2; ++mf)
#pragma unroll
    for (int nf = 0; nf < 2; ++nf) {
      v8f g = accG[mf][nf], u = accU[mf][nf];
#pragma unroll
      for (int i = 0; i < 8; ++i) {
        float gv = g[i], uv = u[i];
        float s  = gv / (1.0f + __expf(-gv));      // silu
        int m = mrow0 + mf * 16 + i + ((lane >> 4) << 3);
        int n = ncol0 + nf * 16 + (lane & 15);
        actE[(size_t)m * DFF_ + n] = f2bf1(s * uv);
      }
    }
}

// --------------------------- GEMM2 (act @ down) -----------------------------
__global__ void __launch_bounds__(256)
gemm2_kernel(const unsigned short* __restrict__ act,
             const float* __restrict__ down,
             float* __restrict__ yd) {
  constexpr int BK = 32;
  const int e    = blockIdx.z;
  const int mt   = blockIdx.x;    // 0..3
  const int nt   = blockIdx.y;    // 0..15 (64-col slab of d=1024)
  const int tid  = threadIdx.x;
  const int lane = tid & 31;
  const int wid  = tid >> 5;
  const int wm   = wid & 3;
  const int wn   = wid >> 2;

  __shared__ unsigned short sA[128 * BK];
  __shared__ unsigned short sB[BK * 64];

  const unsigned short* Abase = act + ((size_t)e * CAP_ + (size_t)mt * 128) * DFF_;
  const float* Bd = down + (size_t)e * DFF_ * D_ + nt * 64;

  const v8f vzero = {0.f, 0.f, 0.f, 0.f, 0.f, 0.f, 0.f, 0.f};
  v8f acc[2][2];
#pragma unroll
  for (int i = 0; i < 2; ++i)
#pragma unroll
    for (int j = 0; j < 2; ++j) acc[i][j] = vzero;

  const int aRow = tid >> 1, aHalf = tid & 1;
  // B loader: 256 threads, each handles TWO adjacent K rows x 4 cols.
  const int bKr   = (tid >> 4) * 2;   // even K row: 0,2,..,30
  const int bN0   = (tid & 15) * 4;   // 0..60
  const int bNb   = bN0 >> 4;
  const int bNlo  = bN0 & 15;         // 0,4,8,12
  const int bLsel = bKr & 16;
  const int bSlot = bKr & 15;

  for (int k0 = 0; k0 < DFF_; k0 += BK) {
    __syncthreads();
#ifdef HAVE_TDM
    if (wid == 0) tdm_load_tile_128x32_bf16(Abase + k0, sA, DFF_);
#else
    {
      const uint4* s = (const uint4*)(Abase + (size_t)aRow * DFF_ + k0 + aHalf * 16);
      uint4 a0 = s[0], a1 = s[1];
      uint4* d = (uint4*)&sA[aRow * BK + aHalf * 16];
      d[0] = a0; d[1] = a1;
    }
#endif
    {
      const float* s0 = Bd + (size_t)(k0 + bKr) * D_ + bN0;
      const float* s1 = s0 + D_;
      float4 r0 = ((const float4*)s0)[0];
      float4 r1 = ((const float4*)s1)[0];
      float v0[4] = {r0.x, r0.y, r0.z, r0.w};
      float v1[4] = {r1.x, r1.y, r1.z, r1.w};
      unsigned short* base = sB + bNb * 512 + bSlot;
#pragma unroll
      for (int j = 0; j < 4; ++j) {
        unsigned pk = pack2bf(v0[j], v1[j]);
        *(unsigned*)&base[(bNlo + j + bLsel) * 16] = pk;
      }
    }
#ifdef HAVE_TDM
    if (wid == 0) __builtin_amdgcn_s_wait_tensorcnt(0);
#endif
    __syncthreads();

    if (k0 + BK < DFF_) {
      __builtin_prefetch(Abase + (size_t)aRow * DFF_ + k0 + BK, 0, 1);
      __builtin_prefetch(Bd + (size_t)(k0 + BK + bKr) * D_ + bN0, 0, 1);
    }

    Frag16 a[2];
#pragma unroll
    for (int mf = 0; mf < 2; ++mf) {
      int r  = wm * 32 + mf * 16 + (lane & 15);
      int kh = lane >> 4;
      a[mf].q[0] = *(const uint4*)&sA[r * BK + kh * 16];
      a[mf].q[1] = *(const uint4*)&sA[r * BK + kh * 16 + 8];
    }
    Frag16 b[2];
#pragma unroll
    for (int nf = 0; nf < 2; ++nf) {
      int nb = wn * 2 + nf;
      const uint4* pb = (const uint4*)&sB[nb * 512 + lane * 16];
      b[nf].q[0] = pb[0]; b[nf].q[1] = pb[1];
    }
#pragma unroll
    for (int mf = 0; mf < 2; ++mf)
#pragma unroll
      for (int nf = 0; nf < 2; ++nf)
        acc[mf][nf] = __builtin_amdgcn_wmma_f32_16x16x32_bf16(
            false, a[mf].v, false, b[nf].v, (short)0, acc[mf][nf], false, false);
  }

  const int mrow0 = mt * 128 + wm * 32;
  const int ncol0 = nt * 64 + wn * 32;
  float* ydE = yd + (size_t)e * CAP_ * D_;
#pragma unroll
  for (int mf = 0; mf < 2; ++mf)
#pragma unroll
    for (int nf = 0; nf < 2; ++nf) {
      v8f v = acc[mf][nf];
#pragma unroll
      for (int i = 0; i < 8; ++i) {
        int m = mrow0 + mf * 16 + i + ((lane >> 4) << 3);
        int n = ncol0 + nf * 16 + (lane & 15);
        ydE[(size_t)m * D_ + n] = v[i];
      }
    }
}

// ------------------------------ combine ------------------------------------
__global__ void combine_kernel(const int* __restrict__ idx,
                               const float* __restrict__ w,
                               const int* __restrict__ pos,
                               const float* __restrict__ yd,
                               float* __restrict__ out) {
  size_t i = (size_t)blockIdx.x * blockDim.x + threadIdx.x;
  int t = (int)(i >> 10);        // / D_
  int c = (int)(i & 1023);       // % D_
  float acc = 0.f;
#pragma unroll
  for (int j = 0; j < K_; ++j) {
    int n = t * K_ + j;
    int p = pos[n];
    if (p < CAP_) {
      int e = idx[n];
      acc += w[n] * yd[((size_t)e * CAP_ + p) * D_ + c];
    }
  }
  out[i] = acc;
}

// ------------------------------ launcher -----------------------------------
extern "C" void kernel_launch(void* const* d_in, const int* in_sizes, int n_in,
                              void* d_out, int out_size, void* d_ws, size_t ws_size,
                              hipStream_t stream) {
  const float* hs  = (const float*)d_in[0];   // hidden_states [T, d] f32
  const int*   idx = (const int*)d_in[1];     // topk_indices  [T, k] i32
  const float* tw  = (const float*)d_in[2];   // topk_weights  [T, k] f32
  const float* gup = (const float*)d_in[3];   // gate_up_proj  [E, d, 2*dff] f32
  const float* dwn = (const float*)d_in[4];   // down_proj     [E, dff, d] f32
  float* out = (float*)d_out;                 // [T, d] f32

  char* ws = (char*)d_ws;
  size_t off = 0;
  auto carve = [&](size_t bytes) {
    void* p = ws + off;
    off = (off + bytes + 255) & ~(size_t)255;
    return p;
  };
  int*            pos = (int*)carve((size_t)N_ * sizeof(int));
  unsigned short* xd  = (unsigned short*)carve((size_t)E_ * CAP_ * D_ * 2);
  unsigned short* act = (unsigned short*)carve((size_t)E_ * CAP_ * DFF_ * 2);
  float*          yd  = (float*)carve((size_t)E_ * CAP_ * D_ * 4);

  route_kernel<<<1, 64, 0, stream>>>(idx, pos);
  zero_xd_kernel<<<(E_ * CAP_ * D_ / 8) / 256, 256, 0, stream>>>((uint4*)xd);
  dispatch_kernel<<<N_, 256, 0, stream>>>(hs, idx, pos, xd);
  gemm1_swiglu_kernel<<<dim3(CAP_ / 128, DFF_ / 64, E_), 256, 0, stream>>>(xd, gup, act);
  gemm2_kernel<<<dim3(CAP_ / 128, D_ / 64, E_), 256, 0, stream>>>(act, dwn, yd);
  combine_kernel<<<(unsigned)(((size_t)T_ * D_) / 256), 256, 0, stream>>>(idx, tw, pos, yd, out);
}
